// GNNPolicy_77378130804777
// MI455X (gfx1250) — matrix operand order repaired and checked by reading, not compiled, not run on hardware
//
#include <hip/hip_runtime.h>

typedef __attribute__((ext_vector_type(16))) _Float16 v16h;
typedef __attribute__((ext_vector_type(8)))  float    v8f;

#define NN 100000
#define NE 1600000
#define LN_EPS 1e-5f

__device__ __forceinline__ v8f wmma16(v16h a, v16h b, v8f c) {
  return __builtin_amdgcn_wmma_f32_16x16x32_f16(false, a, false, b, (short)0, c, false, false);
}

// A-fragment K index for element j, lane-half hs (16-bit A 16x32 layout)
__device__ __forceinline__ int a_kidx(int j, int hs) {
  return (j < 8) ? (j + 8 * hs) : ((j - 8) + 16 + 8 * hs);
}

// Load B fragment from row-major KxN f32 weights (ldn = 64), K-tile kb, N-tile nb.
__device__ __forceinline__ v16h load_bfrag(const float* W, int kb, int nb, int lane) {
  int l16 = lane & 15, hs = lane >> 4;
  v16h b;
#pragma unroll
  for (int j = 0; j < 16; ++j) {
    b[j] = (_Float16)W[(kb + a_kidx(j, hs)) * 64 + nb + l16];
  }
  return b;
}

__device__ __forceinline__ v8f zero_v8f() {
  v8f z = {0.f, 0.f, 0.f, 0.f, 0.f, 0.f, 0.f, 0.f};
  return z;
}

// ---------------------------------------------------------------------------
// zero fill
__global__ void zero_kernel(float* p, int n) {
  int i = blockIdx.x * blockDim.x + threadIdx.x;
  if (i < n) p[i] = 0.f;
}

// ---------------------------------------------------------------------------
// embed first layer: out[n][j] = relu( LN(x[n]) . w1[:,j] + b1[j] ), nf small (4/6)
__global__ void embed1_kernel(const float* __restrict__ x, const float* __restrict__ g,
                              const float* __restrict__ b, const float* __restrict__ w1,
                              const float* __restrict__ b1, float* __restrict__ out, int nf) {
  int idx = blockIdx.x * blockDim.x + threadIdx.x;
  int n = idx >> 6, j = idx & 63;
  float xv[8];
  float mu = 0.f;
  for (int f = 0; f < nf; ++f) { xv[f] = x[n * nf + f]; mu += xv[f]; }
  mu /= (float)nf;
  float var = 0.f;
  for (int f = 0; f < nf; ++f) { float d = xv[f] - mu; var += d * d; }
  var /= (float)nf;
  float rs = rsqrtf(var + LN_EPS);
  float acc = b1[j];
  for (int f = 0; f < nf; ++f) {
    float xh = (xv[f] - mu) * rs * g[f] + b[f];
    acc += xh * w1[f * 64 + j];
  }
  out[n * 64 + j] = fmaxf(acc, 0.f);
}

// ---------------------------------------------------------------------------
// Y[M x 64] = act( X[M x 64] @ W[64 x 64] (+ bias) ), one 16-row tile per wave.
__global__ void node_gemm_kernel(const float* __restrict__ X, const float* __restrict__ W,
                                 const float* __restrict__ bias, float* __restrict__ Y,
                                 int doRelu) {
  int lane = threadIdx.x & 31, wid = threadIdx.x >> 5;
  int rowBase = (blockIdx.x * 2 + wid) * 16;
  int m = lane & 15, hs = lane >> 4;

  v16h Bf[2][4];
#pragma unroll
  for (int kt = 0; kt < 2; ++kt)
#pragma unroll
    for (int nt = 0; nt < 4; ++nt) Bf[kt][nt] = load_bfrag(W, kt * 32, nt * 16, lane);

  v8f acc[4];
#pragma unroll
  for (int nt = 0; nt < 4; ++nt) acc[nt] = zero_v8f();

#pragma unroll
  for (int kt = 0; kt < 2; ++kt) {
    v16h A;
#pragma unroll
    for (int j = 0; j < 16; ++j) {
      int k = kt * 32 + a_kidx(j, hs);
      A[j] = (_Float16)X[(rowBase + m) * 64 + k];
    }
#pragma unroll
    for (int nt = 0; nt < 4; ++nt) acc[nt] = wmma16(A, Bf[kt][nt], acc[nt]);
  }

#pragma unroll
  for (int nt = 0; nt < 4; ++nt) {
    int n = nt * 16 + m;
    float bv = bias ? bias[n] : 0.f;
#pragma unroll
    for (int r = 0; r < 8; ++r) {
      int row = rowBase + r + 8 * hs;
      float v = acc[nt][r] + bv;
      if (doRelu) v = fmaxf(v, 0.f);
      Y[row * 64 + n] = v;
    }
  }
}

// ---------------------------------------------------------------------------
// Fused per-edge message + Wf GEMM + atomic scatter-add.
// h = Lp[dst] + Rp[src] + eb*We ; m = relu(LN(h)) @ Wf + bf ; agg[dst] += m
#define HS 68
__global__ void conv_edge_kernel(const float* __restrict__ Lp, const float* __restrict__ Rp,
                                 const int* __restrict__ dstIdx, const int* __restrict__ srcIdx,
                                 const float* __restrict__ We, const float* __restrict__ ebp,
                                 const float* __restrict__ lnf_g, const float* __restrict__ lnf_b,
                                 const float* __restrict__ Wf, const float* __restrict__ bf,
                                 float* __restrict__ agg) {
  __shared__ float sh[2][16 * HS];
  __shared__ float smu[2][16], srs[2][16];
  __shared__ int sdst[2][16];
  __shared__ float sWe[64], sg[64], sb[64], sbf[64];

  int lane = threadIdx.x & 31, wid = threadIdx.x >> 5;
  if (threadIdx.x < 64) {
    sWe[threadIdx.x] = We[threadIdx.x];
    sg[threadIdx.x] = lnf_g[threadIdx.x];
    sb[threadIdx.x] = lnf_b[threadIdx.x];
    sbf[threadIdx.x] = bf[threadIdx.x];
  }
  __syncthreads();
  float eb = ebp[0];

  v16h Bf[2][4];
#pragma unroll
  for (int kt = 0; kt < 2; ++kt)
#pragma unroll
    for (int nt = 0; nt < 4; ++nt) Bf[kt][nt] = load_bfrag(Wf, kt * 32, nt * 16, lane);

  int waveGlobal = blockIdx.x * 2 + wid;
  int m = lane & 15, hs = lane >> 4;
  int k0 = lane * 2;
  float we0 = eb * sWe[k0], we1 = eb * sWe[k0 + 1];

  for (int t = 0; t < 8; ++t) {
    int tileBase = (waveGlobal * 8 + t) * 16;
    // stage h rows + per-edge LN stats (wave-wide shfl reduce)
    for (int e = 0; e < 16; ++e) {
      int E = tileBase + e;
      int d = dstIdx[E], s = srcIdx[E];
      if (lane == 0) sdst[wid][e] = d;
      float2 a = *(const float2*)(Lp + d * 64 + k0);
      float2 c = *(const float2*)(Rp + s * 64 + k0);
      float h0 = a.x + c.x + we0;
      float h1 = a.y + c.y + we1;
      sh[wid][e * HS + k0] = h0;
      sh[wid][e * HS + k0 + 1] = h1;
      float sum = h0 + h1, sq = h0 * h0 + h1 * h1;
#pragma unroll
      for (int o = 16; o >= 1; o >>= 1) {
        sum += __shfl_xor(sum, o, 32);
        sq += __shfl_xor(sq, o, 32);
      }
      if (lane == 0) {
        float mu = sum * (1.f / 64.f);
        float var = sq * (1.f / 64.f) - mu * mu;
        smu[wid][e] = mu;
        srs[wid][e] = rsqrtf(var + LN_EPS);
      }
    }
    __syncthreads();

    v8f acc[4];
#pragma unroll
    for (int nt = 0; nt < 4; ++nt) acc[nt] = zero_v8f();

    float mu = smu[wid][m], rs = srs[wid][m];
#pragma unroll
    for (int kt = 0; kt < 2; ++kt) {
      v16h A;
#pragma unroll
      for (int j = 0; j < 16; ++j) {
        int k = kt * 32 + a_kidx(j, hs);
        float x = (sh[wid][m * HS + k] - mu) * rs * sg[k] + sb[k];
        A[j] = (_Float16)fmaxf(x, 0.f);
      }
#pragma unroll
      for (int nt = 0; nt < 4; ++nt) acc[nt] = wmma16(A, Bf[kt][nt], acc[nt]);
    }

    // scatter-add messages
#pragma unroll
    for (int nt = 0; nt < 4; ++nt) {
      int n = nt * 16 + m;
      float bv = sbf[n];
#pragma unroll
      for (int r = 0; r < 8; ++r) {
        int row = r + 8 * hs;
        int d = sdst[wid][row];
        atomicAdd(&agg[d * 64 + n], acc[nt][r] + bv);
      }
    }
    __syncthreads();
  }
}

// ---------------------------------------------------------------------------
// Fused conv output: cat = [LN(agg), right]; out = relu(cat@Wo1+bo1)@Wo2 + bo2
#define RS 132
__global__ void conv_out_kernel(const float* __restrict__ agg, const float* __restrict__ right,
                                const float* __restrict__ lnp_g, const float* __restrict__ lnp_b,
                                const float* __restrict__ Wo1, const float* __restrict__ bo1,
                                const float* __restrict__ Wo2, const float* __restrict__ bo2,
                                float* __restrict__ out) {
  __shared__ float sh[2][16 * RS];
  __shared__ float smu[2][16], srs[2][16];
  __shared__ float sg[64], sb[64], sbo1[64], sbo2[64];

  int lane = threadIdx.x & 31, wid = threadIdx.x >> 5;
  if (threadIdx.x < 64) {
    sg[threadIdx.x] = lnp_g[threadIdx.x];
    sb[threadIdx.x] = lnp_b[threadIdx.x];
    sbo1[threadIdx.x] = bo1[threadIdx.x];
    sbo2[threadIdx.x] = bo2[threadIdx.x];
  }
  __syncthreads();

  int nodeBase = (blockIdx.x * 2 + wid) * 16;
  int m = lane & 15, hs = lane >> 4;
  int k0 = lane * 2;

  // stage agg (cols 0..63) and right (cols 64..127), LN stats on agg
  for (int e = 0; e < 16; ++e) {
    int n = nodeBase + e;
    float2 a = *(const float2*)(agg + n * 64 + k0);
    float2 r = *(const float2*)(right + n * 64 + k0);
    sh[wid][e * RS + k0] = a.x;
    sh[wid][e * RS + k0 + 1] = a.y;
    sh[wid][e * RS + 64 + k0] = r.x;
    sh[wid][e * RS + 64 + k0 + 1] = r.y;
    float sum = a.x + a.y, sq = a.x * a.x + a.y * a.y;
#pragma unroll
    for (int o = 16; o >= 1; o >>= 1) {
      sum += __shfl_xor(sum, o, 32);
      sq += __shfl_xor(sq, o, 32);
    }
    if (lane == 0) {
      float mu = sum * (1.f / 64.f);
      float var = sq * (1.f / 64.f) - mu * mu;
      smu[wid][e] = mu;
      srs[wid][e] = rsqrtf(var + LN_EPS);
    }
  }
  __syncthreads();

  // phase 1: t = relu(cat @ Wo1 + bo1), K = 128
  v8f acc[4];
#pragma unroll
  for (int nt = 0; nt < 4; ++nt) acc[nt] = zero_v8f();
  float mu = smu[wid][m], rs = srs[wid][m];
#pragma unroll
  for (int kt = 0; kt < 4; ++kt) {
    v16h A;
#pragma unroll
    for (int j = 0; j < 16; ++j) {
      int k = kt * 32 + a_kidx(j, hs);
      float x;
      if (k < 64) x = (sh[wid][m * RS + k] - mu) * rs * sg[k] + sb[k];
      else        x = sh[wid][m * RS + k];  // right stored at cols 64..127
      A[j] = (_Float16)x;
    }
#pragma unroll
    for (int nt = 0; nt < 4; ++nt) {
      v16h B = load_bfrag(Wo1, kt * 32, nt * 16, lane);
      acc[nt] = wmma16(A, B, acc[nt]);
    }
  }
  __syncthreads();

  // stage t back into LDS (cols 0..63)
#pragma unroll
  for (int nt = 0; nt < 4; ++nt) {
    int n = nt * 16 + m;
#pragma unroll
    for (int r = 0; r < 8; ++r) {
      int row = r + 8 * hs;
      sh[wid][row * RS + n] = fmaxf(acc[nt][r] + sbo1[n], 0.f);
    }
  }
  __syncthreads();

  // phase 2: out = t @ Wo2 + bo2, K = 64
  v8f acc2[4];
#pragma unroll
  for (int nt = 0; nt < 4; ++nt) acc2[nt] = zero_v8f();
#pragma unroll
  for (int kt = 0; kt < 2; ++kt) {
    v16h A;
#pragma unroll
    for (int j = 0; j < 16; ++j) {
      int k = kt * 32 + a_kidx(j, hs);
      A[j] = (_Float16)sh[wid][m * RS + k];
    }
#pragma unroll
    for (int nt = 0; nt < 4; ++nt) {
      v16h B = load_bfrag(Wo2, kt * 32, nt * 16, lane);
      acc2[nt] = wmma16(A, B, acc2[nt]);
    }
  }
#pragma unroll
  for (int nt = 0; nt < 4; ++nt) {
    int n = nt * 16 + m;
#pragma unroll
    for (int r = 0; r < 8; ++r) {
      int row = nodeBase + r + 8 * hs;
      out[row * 64 + n] = acc2[nt][r] + sbo2[n];
    }
  }
}

// ---------------------------------------------------------------------------
// decoder final dot: out[n] = H[n,:] . w2  (H already relu'd), one wave per node
__global__ void dec_dot_kernel(const float* __restrict__ H, const float* __restrict__ w2,
                               float* __restrict__ out) {
  int lane = threadIdx.x & 31, wid = threadIdx.x >> 5;
  int n = blockIdx.x * 8 + wid;
  float s = H[n * 64 + lane] * w2[lane] + H[n * 64 + 32 + lane] * w2[32 + lane];
#pragma unroll
  for (int o = 16; o >= 1; o >>= 1) s += __shfl_xor(s, o, 32);
  if (lane == 0) out[n] = s;
}

// ---------------------------------------------------------------------------
extern "C" void kernel_launch(void* const* d_in, const int* in_sizes, int n_in,
                              void* d_out, int out_size, void* d_ws, size_t ws_size,
                              hipStream_t stream) {
  (void)in_sizes; (void)n_in; (void)out_size; (void)ws_size;
  auto F = [&](int i) { return (const float*)d_in[i]; };

  const float* cons = F(0);
  const int* ei = (const int*)d_in[1];
  const float* vars = F(3);

  // cons_emb: ln_g ln_b w1 b1 w2 b2  (4..9) ; var_emb (10..15); edge_ln g,b (16,17)
  const float *ce_g = F(4), *ce_b = F(5), *ce_w1 = F(6), *ce_b1 = F(7), *ce_w2 = F(8), *ce_b2 = F(9);
  const float *ve_g = F(10), *ve_b = F(11), *ve_w1 = F(12), *ve_b1 = F(13), *ve_w2 = F(14), *ve_b2 = F(15);
  const float* el_b = F(17);

  struct ConvP { const float *Wl, *bl, *We, *Wr, *lng, *lnb, *Wf, *bf, *lpg, *lpb, *Wo1, *bo1, *Wo2, *bo2; };
  ConvP cp[4];
  int base = 18;
  for (int i = 0; i < 4; ++i) {
    cp[i] = {F(base + 0), F(base + 1), F(base + 2),  F(base + 3), F(base + 4),  F(base + 5),
             F(base + 6), F(base + 7), F(base + 8),  F(base + 9), F(base + 10), F(base + 11),
             F(base + 12), F(base + 13)};
    base += 14;
  }
  const float *dw1 = F(74), *db1 = F(75), *dw2 = F(76);

  const int* ei_c = ei;
  const int* ei_v = ei + NE;

  size_t NB = (size_t)NN * 64;
  float* W[7];
  for (int i = 0; i < 7; ++i) W[i] = (float*)d_ws + i * NB;
  float* c = W[0];
  float* v = W[1];
  float* Lp = W[2];
  float* Rp = W[3];
  float* agg = W[4];
  float* convOut[4] = {W[5], W[6], W[0], W[1]};

  // --- embeddings ---
  embed1_kernel<<<25000, 256, 0, stream>>>(cons, ce_g, ce_b, ce_w1, ce_b1, Lp, 4);
  node_gemm_kernel<<<3125, 64, 0, stream>>>(Lp, ce_w2, ce_b2, c, 1);
  embed1_kernel<<<25000, 256, 0, stream>>>(vars, ve_g, ve_b, ve_w1, ve_b1, Lp, 6);
  node_gemm_kernel<<<3125, 64, 0, stream>>>(Lp, ve_w2, ve_b2, v, 1);

  // --- 4 bipartite convolutions ---
  for (int i = 0; i < 4; ++i) {
    bool toC = (i % 2 == 0);
    const float* right = toC ? c : v;
    const float* left = toC ? v : c;
    const int* dst = toC ? ei_c : ei_v;
    const int* src = toC ? ei_v : ei_c;

    node_gemm_kernel<<<3125, 64, 0, stream>>>(right, cp[i].Wl, cp[i].bl, Lp, 0);
    node_gemm_kernel<<<3125, 64, 0, stream>>>(left, cp[i].Wr, nullptr, Rp, 0);
    zero_kernel<<<25000, 256, 0, stream>>>(agg, NN * 64);
    conv_edge_kernel<<<6250, 64, 0, stream>>>(Lp, Rp, dst, src, cp[i].We, el_b, cp[i].lng,
                                              cp[i].lnb, cp[i].Wf, cp[i].bf, agg);
    conv_out_kernel<<<3125, 64, 0, stream>>>(agg, right, cp[i].lpg, cp[i].lpb, cp[i].Wo1,
                                             cp[i].bo1, cp[i].Wo2, cp[i].bo2, convOut[i]);
    if (toC) c = convOut[i]; else v = convOut[i];
  }

  // --- decoder ---
  node_gemm_kernel<<<3125, 64, 0, stream>>>(v, dw1, db1, Lp, 1);
  dec_dot_kernel<<<12500, 256, 0, stream>>>(Lp, dw2, (float*)d_out);
}
